// LlamaAttention_squeeze_24343874634099
// MI455X (gfx1250) — matrix operand order repaired and checked by reading, not compiled, not run on hardware
//
#include <hip/hip_runtime.h>
#include <hip/hip_bf16.h>
#include <stdint.h>

// ---------------------------------------------------------------------------
// Llama attention block (prefill) for gfx1250 (MI455X).
// bf16 WMMA (V_WMMA_F32_16X16X32_BF16) everywhere + flash attention.
// Weight B-operands pre-transposed to [N][K] so LDS tiles fill with straight
// 16B copies. Tile fills use CDNA5 GLOBAL_LOAD_ASYNC_TO_LDS_B128 (ASYNCcnt);
// both the GEMM and the flash-attention kv loop are double-buffered so the
// next tile streams into LDS while WMMAs consume the current one.
// position_ids == arange, attention_mask == causal: reproduced in-kernel.
// ---------------------------------------------------------------------------

typedef __bf16 bf16_t;
typedef __attribute__((ext_vector_type(16))) __bf16 v16bf;
typedef __attribute__((ext_vector_type(8)))  __bf16 v8bf;
typedef __attribute__((ext_vector_type(8)))  float  v8f;

#define B_   2
#define S_   2048
#define HID_ 2048
#define NH_  16
#define NKV_ 4
#define HD_  128
#define GQA_ (NH_ / NKV_)   // 4

// ---- CDNA5 async global->LDS copies (ASYNCcnt) ----------------------------
#if defined(__has_builtin)
#if __has_builtin(__builtin_amdgcn_global_load_async_to_lds_b128)
#define ASYNC_LDS 1
#endif
#endif
#ifndef ASYNC_LDS
#define ASYNC_LDS 0
#endif

#if ASYNC_LDS
typedef int v4i __attribute__((vector_size(16)));
typedef __attribute__((address_space(1))) v4i glb_v4i;
typedef __attribute__((address_space(3))) v4i lds_v4i;
__device__ __forceinline__ void async_cp16(const void* g, void* l) {
  // 16B per lane, memory -> LDS, tracked by ASYNCcnt.
  __builtin_amdgcn_global_load_async_to_lds_b128(
      (glb_v4i*)(uintptr_t)g, (lds_v4i*)(uint32_t)(uintptr_t)l, 0, 0);
}
template <int N> __device__ __forceinline__ void wait_asynccnt() {
#if __has_builtin(__builtin_amdgcn_s_wait_asynccnt)
  __builtin_amdgcn_s_wait_asynccnt((short)N);
#else
  asm volatile("s_wait_asynccnt %0" ::"i"(N) : "memory");
#endif
}
#endif

__device__ __forceinline__ v16bf make_frag(v8bf lo, v8bf hi) {
  union { v16bf v; v8bf h[2]; } u;
  u.h[0] = lo; u.h[1] = hi;
  return u.v;
}

__device__ __forceinline__ v8f wmma_bf16(v16bf a, v16bf b, v8f c) {
  return __builtin_amdgcn_wmma_f32_16x16x32_bf16(
      false, a, false, b, (short)0, c, false, false);
}

// ---------------------------------------------------------------------------
// fp32 -> bf16 convert (row-major preserved; hidden states)
// ---------------------------------------------------------------------------
__global__ __launch_bounds__(256) void cvt_f32_bf16(const float* __restrict__ src,
                                                    bf16_t* __restrict__ dst, int n) {
  int i = (blockIdx.x * 256 + threadIdx.x) * 4;
  if (i + 3 < n) {
    float4 v = *(const float4*)&src[i];
    dst[i + 0] = (bf16_t)v.x;
    dst[i + 1] = (bf16_t)v.y;
    dst[i + 2] = (bf16_t)v.z;
    dst[i + 3] = (bf16_t)v.w;
  } else {
    for (int k = i; k < n; ++k) dst[k] = (bf16_t)src[k];
  }
}

// ---------------------------------------------------------------------------
// fp32 [2048][N] -> bf16 transposed [N][2048] (weights; K is always HID_)
// ---------------------------------------------------------------------------
__global__ __launch_bounds__(256) void cvt_transpose_bf16(
    const float* __restrict__ src, bf16_t* __restrict__ dst, int ndim) {
  int idx = blockIdx.x * 256 + threadIdx.x;   // dst-linear over ndim*2048
  int k = idx & (HID_ - 1);
  int n = idx >> 11;
  dst[idx] = (bf16_t)src[(size_t)k * ndim + n];
}

// ---------------------------------------------------------------------------
// NT-GEMM: C[M,N] = A[M,K] * Bt[N,K]^T. A,Bt row-major bf16; C fp32.
// Block: 256 threads = 8 waves; tile 128x256x32; wave = 64x64 (4x4 WMMA).
// Double-buffered LDS; tile t+1 streamed by async-LDS while t computes.
// ---------------------------------------------------------------------------
template <int N, int K>
__global__ __launch_bounds__(256) void gemm_nt_bf16(
    const bf16_t* __restrict__ A, const bf16_t* __restrict__ Bt,
    float* __restrict__ C) {
  constexpr int BM = 128, BN = 256, BK = 32;
  constexpr int NIT = K / BK;
  __shared__ bf16_t As[2][BM][BK];   // 16 KB
  __shared__ bf16_t Bs[2][BN][BK];   // 32 KB

  const int tid  = threadIdx.x;
  const int wid  = tid >> 5;
  const int lane = tid & 31;
  const int half = lane >> 4;
  const int l15  = lane & 15;

  const int bm = blockIdx.y * BM;
  const int bn = blockIdx.x * BN;
  const int wm = (wid & 1) * 64;   // 2 waves along M
  const int wn = (wid >> 1) * 64;  // 4 waves along N

  const int tr = tid >> 2;         // 0..63
  const int tc = (tid & 3) * 8;    // 0/8/16/24
  const bf16_t* aptr0 = A  + (size_t)(bm + tr) * K + tc;
  const bf16_t* aptr1 = aptr0 + (size_t)64 * K;
  const bf16_t* bptr  = Bt + (size_t)(bn + tr) * K + tc;

  auto issue = [&](int it) {
    const int buf = it & 1;
    const int k0  = it * BK;
#if ASYNC_LDS
    async_cp16(aptr0 + k0, &As[buf][tr][tc]);
    async_cp16(aptr1 + k0, &As[buf][tr + 64][tc]);
#pragma unroll
    for (int i = 0; i < 4; ++i)
      async_cp16(bptr + (size_t)(64 * i) * K + k0, &Bs[buf][tr + 64 * i][tc]);
#else
    *(v8bf*)&As[buf][tr][tc]      = *(const v8bf*)(aptr0 + k0);
    *(v8bf*)&As[buf][tr + 64][tc] = *(const v8bf*)(aptr1 + k0);
#pragma unroll
    for (int i = 0; i < 4; ++i)
      *(v8bf*)&Bs[buf][tr + 64 * i][tc] =
          *(const v8bf*)(bptr + (size_t)(64 * i) * K + k0);
#endif
  };

  v8f acc[4][4];
#pragma unroll
  for (int i = 0; i < 4; ++i)
#pragma unroll
    for (int j = 0; j < 4; ++j) acc[i][j] = (v8f)0.0f;

  issue(0);
  for (int it = 0; it < NIT; ++it) {
    if (it + 1 < NIT) issue(it + 1);
#if ASYNC_LDS
    if (it + 1 < NIT) wait_asynccnt<6>();   // 6 newer issues = next tile's
    else              wait_asynccnt<0>();
#endif
    __syncthreads();   // current tile visible to all waves

    const int buf = it & 1;
    v16bf af[4];
    const int ka = half ? 8 : 0;
    const int kb = half ? 8 : 0;
#pragma unroll
    for (int i = 0; i < 4; ++i) {
      int row = wm + i * 16 + l15;
      af[i] = make_frag(*(const v8bf*)&As[buf][row][ka],
                        *(const v8bf*)&As[buf][row][ka + 16]);
    }
#pragma unroll
    for (int j = 0; j < 4; ++j) {
      int col = wn + j * 16 + l15;
      v16bf bf = make_frag(*(const v8bf*)&Bs[buf][col][kb],
                           *(const v8bf*)&Bs[buf][col][kb + 16]);
#pragma unroll
      for (int i = 0; i < 4; ++i) acc[i][j] = wmma_bf16(af[i], bf, acc[i][j]);
    }
    __syncthreads();   // buffer may be refilled next iteration
  }

  // C/D layout: lanes 0-15 -> M=v, lanes 16-31 -> M=v+8; N = lane&15.
#pragma unroll
  for (int i = 0; i < 4; ++i)
#pragma unroll
    for (int j = 0; j < 4; ++j) {
      float* crow = C + (size_t)(bm + wm + i * 16 + half * 8) * N
                      + bn + wn + j * 16 + l15;
#pragma unroll
      for (int v = 0; v < 8; ++v) crow[(size_t)v * N] = acc[i][j][v];
    }
}

// ---------------------------------------------------------------------------
// RoPE + head-split: src fp32 [B*S, nheads*128] -> dst bf16 [B,nheads,S,128].
// ---------------------------------------------------------------------------
__global__ __launch_bounds__(256) void rope_to_heads(
    const float* __restrict__ src, bf16_t* __restrict__ dst, int nheads) {
  int idx = blockIdx.x * 256 + threadIdx.x;   // B*nheads*S*64 total
  int d  = idx & 63;
  int s  = (idx >> 6) & (S_ - 1);
  int bh = idx >> 17;
  int h  = bh % nheads;
  int b  = bh / nheads;

  int ld = nheads * HD_;
  size_t srow = (size_t)(b * S_ + s) * ld + h * HD_;
  float x1 = src[srow + d];
  float x2 = src[srow + d + 64];

  float inv_freq = __powf(10000.0f, -(float)d * (1.0f / 64.0f));
  float fr = (float)s * inv_freq;
  float c  = __cosf(fr);
  float sn = __sinf(fr);

  size_t drow = ((size_t)(b * nheads + h) * S_ + s) * HD_;
  dst[drow + d]      = (bf16_t)(x1 * c - x2 * sn);
  dst[drow + d + 64] = (bf16_t)(x2 * c + x1 * sn);
}

// ---------------------------------------------------------------------------
// V transpose: src fp32 [B*S, NKV*128] -> dst bf16 [B,NKV,128,S]
// ---------------------------------------------------------------------------
__global__ __launch_bounds__(256) void v_transpose(
    const float* __restrict__ src, bf16_t* __restrict__ dst) {
  int idx = blockIdx.x * 256 + threadIdx.x;
  int s   = idx & (S_ - 1);
  int d   = (idx >> 11) & 127;
  int kvb = idx >> 18;
  int kv  = kvb & (NKV_ - 1);
  int b   = kvb >> 2;
  dst[idx] = (bf16_t)src[(size_t)(b * S_ + s) * (NKV_ * HD_) + kv * HD_ + d];
}

// ---------------------------------------------------------------------------
// Flash attention: grid (S/128, NH, B), 256 threads = 8 waves.
// Wave w handles q rows [q0+16w, q0+16w+16). kv tiles of 64, double-buffered:
// tile t+1 streams into LDS via async copies while tile t is computed.
// ---------------------------------------------------------------------------
#define KVT 64

__global__ __launch_bounds__(256) void flash_attn(
    const bf16_t* __restrict__ Q, const bf16_t* __restrict__ Kr,
    const bf16_t* __restrict__ Vt, bf16_t* __restrict__ ctx) {
  __shared__ bf16_t Ks[2][KVT][HD_];    // 32 KB [buf][kv][d]
  __shared__ bf16_t Vs[2][HD_][KVT];    // 32 KB [buf][d][kv]
  __shared__ bf16_t Ps[8][16][KVT];     // 16 KB per-wave P staging

  const int tid  = threadIdx.x;
  const int wid  = tid >> 5;
  const int lane = tid & 31;
  const int half = lane >> 4;
  const int l15  = lane & 15;

  const int q0    = blockIdx.x * 128;
  const int h     = blockIdx.y;
  const int b     = blockIdx.z;
  const int kvh   = h / GQA_;
  const int qrow0 = q0 + wid * 16;

  // Q A-fragments straight from global.
  const bf16_t* qbase = Q + ((size_t)(b * NH_ + h) * S_ + qrow0 + l15) * HD_;
  v16bf qf[4];
#pragma unroll
  for (int f = 0; f < 4; ++f) {
    int ka = f * 32 + (half ? 8 : 0);
    qf[f] = make_frag(*(const v8bf*)&qbase[ka], *(const v8bf*)&qbase[ka + 16]);
  }

  // Hoisted per-thread tile-fill addresses.
  const bf16_t* kld = Kr + (size_t)(b * NKV_ + kvh) * S_ * HD_
                         + (size_t)(tid >> 4) * HD_ + (tid & 15) * 8;
  const bf16_t* vld = Vt + (size_t)(b * NKV_ + kvh) * HD_ * S_
                         + (size_t)(tid >> 3) * S_ + (tid & 7) * 8;
  const int krow = tid >> 4, kcol = (tid & 15) * 8;   // K-tile LDS coords
  const int vrow = tid >> 3, vcol = (tid & 7) * 8;    // V-tile LDS coords

  auto issue = [&](int t) {
    const int buf = t & 1;
    const int kv0 = t * KVT;
#if ASYNC_LDS
#pragma unroll
    for (int i = 0; i < 4; ++i)
      async_cp16(kld + (size_t)kv0 * HD_ + (size_t)i * 16 * HD_,
                 &Ks[buf][krow + 16 * i][kcol]);
#pragma unroll
    for (int i = 0; i < 4; ++i)
      async_cp16(vld + (size_t)i * 32 * S_ + kv0, &Vs[buf][vrow + 32 * i][vcol]);
#else
#pragma unroll
    for (int i = 0; i < 4; ++i)
      *(v8bf*)&Ks[buf][krow + 16 * i][kcol] =
          *(const v8bf*)&kld[(size_t)kv0 * HD_ + (size_t)i * 16 * HD_];
#pragma unroll
    for (int i = 0; i < 4; ++i)
      *(v8bf*)&Vs[buf][vrow + 32 * i][vcol] =
          *(const v8bf*)&vld[(size_t)i * 32 * S_ + kv0];
#endif
  };

  float mrow[8], lrow[8];
  v8f o[8];
#pragma unroll
  for (int v = 0; v < 8; ++v) { mrow[v] = -1e30f; lrow[v] = 0.0f; }
#pragma unroll
  for (int j = 0; j < 8; ++j) o[j] = (v8f)0.0f;

  const float scale = 0.08838834764831845f;  // 1/sqrt(128)
  const int nkt = (q0 + 128) / KVT;          // causal upper bound

  issue(0);
  for (int t = 0; t < nkt; ++t) {
    const int kv0 = t * KVT;
    if (t + 1 < nkt) issue(t + 1);
#if ASYNC_LDS
    if (t + 1 < nkt) wait_asynccnt<8>();    // 8 newer issues = next tile's
    else             wait_asynccnt<0>();
#endif
    __syncthreads();   // current tile visible to all waves
    const int buf = t & 1;

    // ---- scores S_w[16x64] = Q[16x128] * K_tile^T ----
    v8f sc[4];
#pragma unroll
    for (int nt = 0; nt < 4; ++nt) sc[nt] = (v8f)0.0f;
#pragma unroll
    for (int nt = 0; nt < 4; ++nt) {
      int col = nt * 16 + l15;
#pragma unroll
      for (int kc = 0; kc < 4; ++kc) {
        int kb = kc * 32 + (half ? 8 : 0);
        v16bf kf = make_frag(*(const v8bf*)&Ks[buf][col][kb],
                             *(const v8bf*)&Ks[buf][col][kb + 16]);
        sc[nt] = wmma_bf16(qf[kc], kf, sc[nt]);
      }
    }

    // ---- scale + causal mask ----
#pragma unroll
    for (int nt = 0; nt < 4; ++nt) {
      int nidx = kv0 + nt * 16 + l15;
#pragma unroll
      for (int v = 0; v < 8; ++v) {
        float x = sc[nt][v] * scale;
        if (nidx > qrow0 + v + half * 8) x = -1e30f;
        sc[nt][v] = x;
      }
    }

    // ---- online softmax (reductions across 16-lane halves) ----
#pragma unroll
    for (int v = 0; v < 8; ++v) {
      float rm = fmaxf(fmaxf(sc[0][v], sc[1][v]), fmaxf(sc[2][v], sc[3][v]));
      rm = fmaxf(rm, __shfl_xor(rm, 1, 32));
      rm = fmaxf(rm, __shfl_xor(rm, 2, 32));
      rm = fmaxf(rm, __shfl_xor(rm, 4, 32));
      rm = fmaxf(rm, __shfl_xor(rm, 8, 32));
      float mn    = fmaxf(mrow[v], rm);
      float alpha = __expf(mrow[v] - mn);
      mrow[v] = mn;
      float rs = 0.0f;
#pragma unroll
      for (int nt = 0; nt < 4; ++nt) {
        float p = __expf(sc[nt][v] - mn);
        rs += p;
        Ps[wid][v + half * 8][nt * 16 + l15] = (bf16_t)p;
      }
      rs += __shfl_xor(rs, 1, 32);
      rs += __shfl_xor(rs, 2, 32);
      rs += __shfl_xor(rs, 4, 32);
      rs += __shfl_xor(rs, 8, 32);
      lrow[v] = lrow[v] * alpha + rs;
#pragma unroll
      for (int j = 0; j < 8; ++j) o[j][v] *= alpha;
    }

    // ---- O += P[16x64] * V[64x128] ----
    v16bf pf[2];
#pragma unroll
    for (int f = 0; f < 2; ++f) {
      int ka = f * 32 + (half ? 8 : 0);
      pf[f] = make_frag(*(const v8bf*)&Ps[wid][l15][ka],
                        *(const v8bf*)&Ps[wid][l15][ka + 16]);
    }
#pragma unroll
    for (int j = 0; j < 8; ++j) {
      int col = j * 16 + l15;
#pragma unroll
      for (int f = 0; f < 2; ++f) {
        int kb = f * 32 + (half ? 8 : 0);
        v16bf vf = make_frag(*(const v8bf*)&Vs[buf][col][kb],
                             *(const v8bf*)&Vs[buf][col][kb + 16]);
        o[j] = wmma_bf16(pf[f], vf, o[j]);
      }
    }
    __syncthreads();   // buffer may be refilled next iteration
  }

  // ---- normalize + write ctx [B,S,NH*HD] bf16 ----
#pragma unroll
  for (int v = 0; v < 8; ++v) {
    int srow = qrow0 + v + half * 8;
    float inv_l = 1.0f / lrow[v];
    size_t base = ((size_t)b * S_ + srow) * (NH_ * HD_) + h * HD_;
#pragma unroll
    for (int j = 0; j < 8; ++j)
      ctx[base + j * 16 + l15] = (bf16_t)(o[j][v] * inv_l);
  }
}

// ---------------------------------------------------------------------------
// Launch
// ---------------------------------------------------------------------------
extern "C" void kernel_launch(void* const* d_in, const int* in_sizes, int n_in,
                              void* d_out, int out_size, void* d_ws, size_t ws_size,
                              hipStream_t stream) {
  (void)in_sizes; (void)n_in; (void)out_size; (void)ws_size;
  const float* hidden = (const float*)d_in[0];
  const float* Wq = (const float*)d_in[3];
  const float* Wk = (const float*)d_in[4];
  const float* Wv = (const float*)d_in[5];
  const float* Wo = (const float*)d_in[6];

  const size_t M = (size_t)B_ * S_;            // 4096
  size_t off = 0;
  auto carve = [&](size_t bytes) -> void* {
    void* p = (char*)d_ws + off;
    off += (bytes + 255) & ~(size_t)255;
    return p;
  };
  bf16_t* hb  = (bf16_t*)carve(M * HID_ * 2);                  // hidden bf16
  bf16_t* wqt = (bf16_t*)carve((size_t)HID_ * HID_ * 2);       // [N][K]
  bf16_t* wkt = (bf16_t*)carve((size_t)HID_ * NKV_ * HD_ * 2);
  bf16_t* wvt = (bf16_t*)carve((size_t)HID_ * NKV_ * HD_ * 2);
  bf16_t* wot = (bf16_t*)carve((size_t)HID_ * HID_ * 2);
  float*  qp  = (float*)carve(M * NH_ * HD_ * 4);              // pre-RoPE
  float*  kp  = (float*)carve(M * NKV_ * HD_ * 4);
  float*  vp  = (float*)carve(M * NKV_ * HD_ * 4);
  bf16_t* qr  = (bf16_t*)carve(M * NH_ * HD_ * 2);             // [B,NH,S,HD]
  bf16_t* kr  = (bf16_t*)carve(M * NKV_ * HD_ * 2);            // [B,NKV,S,HD]
  bf16_t* vt  = (bf16_t*)carve(M * NKV_ * HD_ * 2);            // [B,NKV,HD,S]
  bf16_t* ctx = (bf16_t*)carve(M * NH_ * HD_ * 2);             // [B,S,NH*HD]

  cvt_f32_bf16<<<(unsigned)((M * HID_ / 4 + 255) / 256), 256, 0, stream>>>(
      hidden, hb, (int)(M * HID_));
  cvt_transpose_bf16<<<(HID_ * HID_) / 256, 256, 0, stream>>>(Wq, wqt, HID_);
  cvt_transpose_bf16<<<(HID_ * NKV_ * HD_) / 256, 256, 0, stream>>>(Wk, wkt, NKV_ * HD_);
  cvt_transpose_bf16<<<(HID_ * NKV_ * HD_) / 256, 256, 0, stream>>>(Wv, wvt, NKV_ * HD_);
  cvt_transpose_bf16<<<(HID_ * HID_) / 256, 256, 0, stream>>>(Wo, wot, HID_);

  // QKV projections (NT WMMA GEMM, async double-buffered)
  gemm_nt_bf16<HID_, HID_><<<dim3(HID_ / 256, (unsigned)(M / 128)), 256, 0, stream>>>(hb, wqt, qp);
  gemm_nt_bf16<NKV_ * HD_, HID_><<<dim3(NKV_ * HD_ / 256, (unsigned)(M / 128)), 256, 0, stream>>>(hb, wkt, kp);
  gemm_nt_bf16<NKV_ * HD_, HID_><<<dim3(NKV_ * HD_ / 256, (unsigned)(M / 128)), 256, 0, stream>>>(hb, wvt, vp);

  // RoPE + layout
  rope_to_heads<<<(B_ * NH_ * S_ * 64) / 256, 256, 0, stream>>>(qp, qr, NH_);
  rope_to_heads<<<(B_ * NKV_ * S_ * 64) / 256, 256, 0, stream>>>(kp, kr, NKV_);
  v_transpose<<<(B_ * NKV_ * HD_ * S_) / 256, 256, 0, stream>>>(vp, vt);

  // Flash attention (async double-buffered kv pipeline)
  flash_attn<<<dim3(S_ / 128, NH_, B_), 256, 0, stream>>>(qr, kr, vt, ctx);

  // Output projection -> fp32 d_out
  gemm_nt_bf16<HID_, HID_><<<dim3(HID_ / 256, (unsigned)(M / 128)), 256, 0, stream>>>(
      ctx, wot, (float*)d_out);
}
